// SparseMoE_46523085750482
// MI455X (gfx1250) — matrix operand order repaired and checked by reading, not compiled, not run on hardware
//
#include <hip/hip_runtime.h>

// Problem sizes (compile-time, from reference): B=4,S=2048 -> N=8192 tokens,
// D=1024, H=2048, E=8, K=2.
#define NTOK 8192
#define DDIM 1024
#define HDIM 2048
#define NEXP 8

typedef unsigned short u16;
typedef unsigned int u32;
typedef __attribute__((ext_vector_type(16))) __bf16 v16bf;
typedef __attribute__((ext_vector_type(2)))  __bf16 v2bf;
typedef __attribute__((ext_vector_type(8)))  float  v8f;
typedef __attribute__((ext_vector_type(2)))  unsigned int u32x2;
typedef __attribute__((ext_vector_type(4)))  unsigned int u32x4;
typedef __attribute__((ext_vector_type(4)))  int v4i;

union FragAB { v16bf v; u32x4 q[2]; };

// Direct global->LDS async staging (CDNA5): no VGPR round-trip, ASYNCcnt
// tracked. Builtin signature (from hipcc diagnostic): (v4i AS1*, v4i AS3*,
// imm, imm). Guarded so absence falls back to b128 copies.
#if __has_builtin(__builtin_amdgcn_global_load_async_to_lds_b128) && \
    __has_builtin(__builtin_amdgcn_s_wait_asynccnt)
#define ASYNC_LDS 1
#else
#define ASYNC_LDS 0
#endif

// Copy 16 bf16 (32B) from global to LDS, per thread.
__device__ __forceinline__ void stage32(const u16* g, u16* l) {
#if ASYNC_LDS
  __builtin_amdgcn_global_load_async_to_lds_b128(
      (__attribute__((address_space(1))) v4i*)g,
      (__attribute__((address_space(3))) v4i*)l, 0, 0);
  __builtin_amdgcn_global_load_async_to_lds_b128(
      (__attribute__((address_space(1))) v4i*)(g + 8),
      (__attribute__((address_space(3))) v4i*)(l + 8), 0, 0);
#else
  const u32x4* s = (const u32x4*)g;
  u32x4* d = (u32x4*)l;
  d[0] = s[0]; d[1] = s[1];
#endif
}

__device__ __forceinline__ void stage_sync() {
#if ASYNC_LDS
  __builtin_amdgcn_s_wait_asynccnt(0);
#endif
  __syncthreads();
}

__device__ __forceinline__ u16 f2bf(float f) {
  union { float f; u32 u; } c; c.f = f;
  u32 u = c.u;
  return (u16)((u + 0x7FFFu + ((u >> 16) & 1u)) >> 16);  // RNE
}

__device__ __forceinline__ u32 f2bf_pk(float lo, float hi) {
#if __has_builtin(__builtin_amdgcn_cvt_pk_bf16_f32)
  v2bf r = __builtin_amdgcn_cvt_pk_bf16_f32(lo, hi);
  return __builtin_bit_cast(u32, r);
#else
  union { float f; u32 u; } a, b; a.f = lo; b.f = hi;
  u32 rl = (a.u + 0x7FFFu + ((a.u >> 16) & 1u)) >> 16;
  u32 rh = (b.u + 0x7FFFu + ((b.u >> 16) & 1u)) & 0xFFFF0000u;
  return rl | rh;
#endif
}

// ---------------------------------------------------------------------------
// Zero output + per-expert atomic counters (graph replays don't re-zero ws).
// ---------------------------------------------------------------------------
__global__ __launch_bounds__(256) void moe_zero(float* __restrict__ out, int n,
                                                int* __restrict__ cnt) {
  int i = blockIdx.x * 256 + threadIdx.x;
  if (i < n) out[i] = 0.f;
  if (i < NEXP) cnt[i] = 0;
}

// ---------------------------------------------------------------------------
// Router + x->bf16: one block per token. logits = x@Wg + bg (one wave per
// expert), top-2 + softmax, atomic compaction of (token, gate) per expert,
// and a bf16 copy of this token's x row (converted exactly once here).
// ---------------------------------------------------------------------------
__global__ __launch_bounds__(256) void moe_router(
    const float* __restrict__ x, const float* __restrict__ Wg,
    const float* __restrict__ bg, int* __restrict__ cnt,
    int* __restrict__ idxb, float* __restrict__ gateb,
    u16* __restrict__ xbf) {
  __shared__ float xrow[DDIM];
  __shared__ float slog[NEXP];
  const int t = blockIdx.x;
  const int tid = threadIdx.x;
  const float* xp = x + (size_t)t * DDIM;
  for (int i = tid; i < DDIM; i += 256) xrow[i] = xp[i];
  __syncthreads();
  {
    int c4 = tid << 2;                     // DDIM/4 == 256 == blockDim
    u32x2 p = { f2bf_pk(xrow[c4], xrow[c4 + 1]),
                f2bf_pk(xrow[c4 + 2], xrow[c4 + 3]) };
    *(u32x2*)(xbf + (size_t)t * DDIM + c4) = p;
  }
  const int wid = tid >> 5, lane = tid & 31;
  float p = 0.f;
  for (int k = lane; k < DDIM; k += 32) p += xrow[k] * Wg[k * NEXP + wid];
  for (int off = 16; off > 0; off >>= 1) p += __shfl_xor(p, off, 32);
  if (lane == 0) slog[wid] = p + bg[wid];
  __syncthreads();
  if (tid == 0) {
    float v0 = slog[0]; int i0 = 0;
    for (int e = 1; e < NEXP; ++e)
      if (slog[e] > v0) { v0 = slog[e]; i0 = e; }        // ties -> lowest idx
    float v1 = -3.4e38f; int i1 = 0;
    for (int e = 0; e < NEXP; ++e) {
      if (e == i0) continue;
      if (slog[e] > v1) { v1 = slog[e]; i1 = e; }
    }
    const float tt = __expf(v1 - v0);                    // <= 1, stable
    const float inv = __builtin_amdgcn_rcpf(1.f + tt);
    int s0 = atomicAdd(&cnt[i0], 1);
    idxb[i0 * NTOK + s0] = t; gateb[i0 * NTOK + s0] = inv;
    int s1 = atomicAdd(&cnt[i1], 1);
    idxb[i1 * NTOK + s1] = t; gateb[i1 * NTOK + s1] = tt * inv;
  }
}

// ---------------------------------------------------------------------------
// Per-expert weight pre-pass: fp32 [K][N] -> bf16 [N][K] (transposed) via an
// LDS tile so both global sides stay coalesced; each element converted ONCE.
// 512 (64x64) tiles per matrix, 3 matrices -> grid 1536.
// ---------------------------------------------------------------------------
__global__ __launch_bounds__(256) void moe_cvt_w(
    const float* __restrict__ w1, const float* __restrict__ w3,
    const float* __restrict__ w2, u16* __restrict__ w1T,
    u16* __restrict__ w3T, u16* __restrict__ w2T, int e) {
  __shared__ __align__(16) u16 tl[64 * 72];   // [n][k], stride 72
  const int tile = blockIdx.x;
  const int mat = tile >> 9;                  // 0:w1 1:w3 2:w2
  const int t = tile & 511;
  const float* src; u16* dst; int Kd, N;
  if (mat == 0)      { src = w1 + (size_t)e * DDIM * HDIM; dst = w1T; Kd = DDIM; N = HDIM; }
  else if (mat == 1) { src = w3 + (size_t)e * DDIM * HDIM; dst = w3T; Kd = DDIM; N = HDIM; }
  else               { src = w2 + (size_t)e * HDIM * DDIM; dst = w2T; Kd = HDIM; N = DDIM; }
  const int tn = N >> 6;
  const int k0 = (t / tn) << 6, n0 = (t % tn) << 6;
  const int tid = threadIdx.x;
#pragma unroll
  for (int i = 0; i < 4; ++i) {
    int id = tid + i * 256;
    int k = id >> 4, n4 = (id & 15) << 2;
    const float4 f = *(const float4*)(src + (size_t)(k0 + k) * N + n0 + n4);
    tl[(n4 + 0) * 72 + k] = f2bf(f.x);
    tl[(n4 + 1) * 72 + k] = f2bf(f.y);
    tl[(n4 + 2) * 72 + k] = f2bf(f.z);
    tl[(n4 + 3) * 72 + k] = f2bf(f.w);
  }
  __syncthreads();
  {
    int n = tid >> 2, c16 = (tid & 3) << 4;
    u32x4* d = (u32x4*)(dst + (size_t)(n0 + n) * Kd + k0 + c16);
    const u32x4* s = (const u32x4*)(&tl[n * 72 + c16]);
    d[0] = s[0]; d[1] = s[1];
  }
}

// LDS row stride (bf16 elems): 144B rows -> 16B aligned, staggered banks.
#define BSTR 72
#define TILE_E (64 * BSTR)   // elems per 64x64 bf16 tile buffer

// ---------------------------------------------------------------------------
// FFN1 (per expert): h = silu(x@w1) * (x@w3) on gathered tokens. All-bf16,
// double-buffered LDS staging (async global->LDS), one barrier per K-step.
// Block 64(M) x 64(N), K-step 64; 8 waves = 4(M) x 2(N); each wave 16x32 out
// for both paths = 8 WMMA / K-step.
// ---------------------------------------------------------------------------
__global__ __launch_bounds__(256) void moe_ffn1(
    const u16* __restrict__ xbf, const u16* __restrict__ w1T,
    const u16* __restrict__ w3T, const int* __restrict__ cnt,
    const int* __restrict__ idxb, u16* __restrict__ hbuf, int e) {
  __shared__ __align__(16) u16 smem[2][3 * TILE_E];  // [buf][xA|b1|b3]
  __shared__ int sidx[64];

  const int count = cnt[e];
  const int m0 = blockIdx.y * 64;
  if (m0 >= count) return;                      // uniform exit (EXEC intact)
  const int rows = (count - m0 < 64) ? (count - m0) : 64;
  const int n0 = blockIdx.x * 64;
  const int tid = threadIdx.x;
  if (tid < 64) sidx[tid] = (m0 + tid < count) ? idxb[e * NTOK + m0 + tid] : 0;
  __syncthreads();

  const int wid = tid >> 5, lane = tid & 31;
  const int hf = lane >> 4, lm = lane & 15;
  const int wm = wid >> 1, wn = wid & 1;
  const int r = tid >> 2, c16 = (tid & 3) << 4;   // staging map: 32B/thread

  v8f a10 = {0.f,0.f,0.f,0.f,0.f,0.f,0.f,0.f};
  v8f a11 = a10, a30 = a10, a31 = a10;

  const u16* ga = xbf + (size_t)sidx[r] * DDIM + c16;
  const u16* g1 = w1T + (size_t)(n0 + r) * DDIM + c16;
  const u16* g3 = w3T + (size_t)(n0 + r) * DDIM + c16;
  const int loff = r * BSTR + c16;

  // Prologue: stage K-step 0 into buffer 0.
  stage32(ga, &smem[0][loff]);
  stage32(g1, &smem[0][TILE_E + loff]);
  stage32(g3, &smem[0][2 * TILE_E + loff]);
  stage_sync();

  for (int kt = 0; kt < DDIM; kt += 64) {
    const int p = (kt >> 6) & 1;
    if (kt + 64 < DDIM) {                       // stage next into other buffer
      stage32(ga + kt + 64, &smem[p ^ 1][loff]);
      stage32(g1 + kt + 64, &smem[p ^ 1][TILE_E + loff]);
      stage32(g3 + kt + 64, &smem[p ^ 1][2 * TILE_E + loff]);
    }
    const u16* xA = &smem[p][0];
    const u16* b1 = &smem[p][TILE_E];
    const u16* b3 = &smem[p][2 * TILE_E];

#pragma unroll
    for (int ks = 0; ks < 2; ++ks) {            // two 16x16x32 sub-steps
      FragAB a, c0, c1, d0, d1;
      const u16* ap = &xA[(wm * 16 + lm) * BSTR + ks * 32 + hf * 16];
      a.q[0] = *(const u32x4*)(ap);
      a.q[1] = *(const u32x4*)(ap + 8);
      const u16* bp;
      bp = &b1[(wn * 32 + lm) * BSTR + ks * 32 + hf * 16];
      c0.q[0] = *(const u32x4*)(bp); c0.q[1] = *(const u32x4*)(bp + 8);
      bp = &b1[(wn * 32 + 16 + lm) * BSTR + ks * 32 + hf * 16];
      c1.q[0] = *(const u32x4*)(bp); c1.q[1] = *(const u32x4*)(bp + 8);
      bp = &b3[(wn * 32 + lm) * BSTR + ks * 32 + hf * 16];
      d0.q[0] = *(const u32x4*)(bp); d0.q[1] = *(const u32x4*)(bp + 8);
      bp = &b3[(wn * 32 + 16 + lm) * BSTR + ks * 32 + hf * 16];
      d1.q[0] = *(const u32x4*)(bp); d1.q[1] = *(const u32x4*)(bp + 8);
      a10 = __builtin_amdgcn_wmma_f32_16x16x32_bf16(false, a.v, false, c0.v,
                                                    (short)0, a10, false, false);
      a11 = __builtin_amdgcn_wmma_f32_16x16x32_bf16(false, a.v, false, c1.v,
                                                    (short)0, a11, false, false);
      a30 = __builtin_amdgcn_wmma_f32_16x16x32_bf16(false, a.v, false, d0.v,
                                                    (short)0, a30, false, false);
      a31 = __builtin_amdgcn_wmma_f32_16x16x32_bf16(false, a.v, false, d1.v,
                                                    (short)0, a31, false, false);
    }
    stage_sync();                               // next buffer staged; one
  }                                             // barrier per K-step

  // Fused SwiGLU epilogue; C/D layout: M = v + 8*hf, N = lm.
#pragma unroll
  for (int v = 0; v < 8; ++v) {
    int rr = wm * 16 + v + 8 * hf;
    if (rr < rows) {
      size_t rowb = (size_t)(m0 + rr) * HDIM + n0 + wn * 32 + lm;
      float x1 = a10[v];
      float h0 = x1 * __builtin_amdgcn_rcpf(1.f + __expf(-x1)) * a30[v];
      hbuf[rowb] = f2bf(h0);
      float x2 = a11[v];
      float h1 = x2 * __builtin_amdgcn_rcpf(1.f + __expf(-x2)) * a31[v];
      hbuf[rowb + 16] = f2bf(h1);
    }
  }
}

// ---------------------------------------------------------------------------
// FFN2 (per expert): y = h@w2 ; out[token,:] += gate * y (race-free: each
// token appears once per expert; experts run sequentially on the stream).
// ---------------------------------------------------------------------------
__global__ __launch_bounds__(256) void moe_ffn2(
    const u16* __restrict__ hbuf, const u16* __restrict__ w2T,
    const int* __restrict__ cnt, const int* __restrict__ idxb,
    const float* __restrict__ gateb, float* __restrict__ out, int e) {
  __shared__ __align__(16) u16 smem[2][2 * TILE_E];  // [buf][hA|wB]
  __shared__ int sidx[64];
  __shared__ float sg[64];

  const int count = cnt[e];
  const int m0 = blockIdx.y * 64;
  if (m0 >= count) return;
  const int rows = (count - m0 < 64) ? (count - m0) : 64;
  const int n0 = blockIdx.x * 64;  // over D
  const int tid = threadIdx.x;
  if (tid < 64) {
    int ok = (m0 + tid < count);
    sidx[tid] = ok ? idxb[e * NTOK + m0 + tid] : 0;
    sg[tid]   = ok ? gateb[e * NTOK + m0 + tid] : 0.f;
  }
  __syncthreads();

  const int wid = tid >> 5, lane = tid & 31;
  const int hf = lane >> 4, lm = lane & 15;
  const int wm = wid >> 1, wn = wid & 1;
  const int r = tid >> 2, c16 = (tid & 3) << 4;

  v8f ac0 = {0.f,0.f,0.f,0.f,0.f,0.f,0.f,0.f};
  v8f ac1 = ac0;

  const u16* ga = hbuf + (size_t)(m0 + r) * HDIM + c16;
  const u16* gw = w2T + (size_t)(n0 + r) * HDIM + c16;
  const int loff = r * BSTR + c16;

  stage32(ga, &smem[0][loff]);
  stage32(gw, &smem[0][TILE_E + loff]);
  stage_sync();

  for (int kt = 0; kt < HDIM; kt += 64) {
    const int p = (kt >> 6) & 1;
    if (kt + 64 < HDIM) {
      stage32(ga + kt + 64, &smem[p ^ 1][loff]);
      stage32(gw + kt + 64, &smem[p ^ 1][TILE_E + loff]);
    }
    const u16* hA = &smem[p][0];
    const u16* wB = &smem[p][TILE_E];

#pragma unroll
    for (int ks = 0; ks < 2; ++ks) {
      FragAB a, b0, b1f;
      const u16* ap = &hA[(wm * 16 + lm) * BSTR + ks * 32 + hf * 16];
      a.q[0] = *(const u32x4*)(ap);
      a.q[1] = *(const u32x4*)(ap + 8);
      const u16* bp;
      bp = &wB[(wn * 32 + lm) * BSTR + ks * 32 + hf * 16];
      b0.q[0] = *(const u32x4*)(bp); b0.q[1] = *(const u32x4*)(bp + 8);
      bp = &wB[(wn * 32 + 16 + lm) * BSTR + ks * 32 + hf * 16];
      b1f.q[0] = *(const u32x4*)(bp); b1f.q[1] = *(const u32x4*)(bp + 8);
      ac0 = __builtin_amdgcn_wmma_f32_16x16x32_bf16(false, a.v, false, b0.v,
                                                    (short)0, ac0, false, false);
      ac1 = __builtin_amdgcn_wmma_f32_16x16x32_bf16(false, a.v, false, b1f.v,
                                                    (short)0, ac1, false, false);
    }
    stage_sync();
  }

#pragma unroll
  for (int v = 0; v < 8; ++v) {
    int rr = wm * 16 + v + 8 * hf;
    if (rr < rows) {
      int tok = sidx[rr];
      float g = sg[rr];
      size_t o = (size_t)tok * DDIM + n0 + wn * 32 + lm;
      out[o] += g * ac0[v];
      out[o + 16] += g * ac1[v];
    }
  }
}

// ---------------------------------------------------------------------------
extern "C" void kernel_launch(void* const* d_in, const int* in_sizes, int n_in,
                              void* d_out, int out_size, void* d_ws,
                              size_t ws_size, hipStream_t stream) {
  const float* x  = (const float*)d_in[0];
  const float* Wg = (const float*)d_in[1];
  const float* bg = (const float*)d_in[2];
  const float* w1 = (const float*)d_in[3];
  const float* w3 = (const float*)d_in[4];
  const float* w2 = (const float*)d_in[5];
  float* out = (float*)d_out;

  // Workspace (~60.5 MB):
  //   cnt 256B | idxb 256KB | gateb 256KB | xbf 16MB |
  //   w1T/w3T/w2T 4MB each (per-expert, reused) | hbuf 32MB
  char* ws = (char*)d_ws;
  int*   cnt   = (int*)ws;
  int*   idxb  = (int*)(ws + 256);
  float* gateb = (float*)(ws + 256 + (size_t)NEXP * NTOK * 4);
  u16*   xbf   = (u16*)(ws + 256 + (size_t)2 * NEXP * NTOK * 4);
  u16*   w1T   = xbf + (size_t)NTOK * DDIM;
  u16*   w3T   = w1T + (size_t)HDIM * DDIM;
  u16*   w2T   = w3T + (size_t)HDIM * DDIM;
  u16*   hbuf  = w2T + (size_t)DDIM * HDIM;

  moe_zero<<<(out_size + 255) / 256, 256, 0, stream>>>(out, out_size, cnt);
  moe_router<<<NTOK, 256, 0, stream>>>(x, Wg, bg, cnt, idxb, gateb, xbf);
  for (int e = 0; e < NEXP; ++e) {
    moe_cvt_w<<<1536, 256, 0, stream>>>(w1, w3, w2, w1T, w3T, w2T, e);
    moe_ffn1<<<dim3(HDIM / 64, NTOK / 64), 256, 0, stream>>>(
        xbf, w1T, w3T, cnt, idxb, hbuf, e);
    moe_ffn2<<<dim3(DDIM / 64, NTOK / 64), 256, 0, stream>>>(
        hbuf, w2T, cnt, idxb, gateb, out, e);
  }
}
